// Transformer_1434519077548
// MI455X (gfx1250) — matrix operand order repaired and checked by reading, not compiled, not run on hardware
//
#include <hip/hip_runtime.h>
#include <hip/hip_bf16.h>

// ---------------------------------------------------------------------------
// Types / constants
// ---------------------------------------------------------------------------
typedef __attribute__((ext_vector_type(16))) _Float16 v16h;
typedef __attribute__((ext_vector_type(8)))  float    v8f;
typedef __attribute__((ext_vector_type(4)))  int      v4i;

#define BQ  512   // batch
#define SQ  64    // seq
#define UQ  384   // model dim
#define HQ  8     // heads
#define DQ  48    // head dim
#define MQ  (BQ*SQ)        // 32768 rows
#define MLPQ (2*UQ)        // 768
#define NCQ 250

#define BM 64
#define BN 64
#define BK 64              // two 16x16x32 k-substeps per staged tile
#define LDT (BK + 8)       // LDS row stride in halves (144B, 16B aligned)

enum { EPI_F16 = 0, EPI_GELU_F16 = 1, EPI_STREAM = 2, EPI_POS = 3,
       EPI_QKV = 4, EPI_RES_H = 5, EPI_RES_F = 6 };

struct EpiArgs {
    const float* bias;           // [N] or null
    float*       outf;           // f32 output
    _Float16*    outh;           // f16 output
    const float* wsm;  int widx; // stream softmax weights
    const unsigned char* flag;   // per-row empty flag
    const float* empty;          // (S,U)
    const int*   nidx;           // per-row pos index
    const float* pos;            // (S+1,U)
    const _Float16* resh;        // f16 residual
    const float*    resf;        // f32 residual
    int firstStream;
};

__device__ __forceinline__ float gelu_exact(float x) {
    return 0.5f * x * (1.0f + erff(x * 0.7071067811865476f));
}

// ---------------------------------------------------------------------------
// CDNA5 async global->LDS copy (GLOBAL_LOAD_ASYNC_TO_LDS_B128, ASYNCcnt).
// Bypasses VGPRs entirely: no staging registers, no spills.
// Param types per hipcc diagnostic: (int4 AS1*, int4 AS3*, imm offset, imm cpol).
// ---------------------------------------------------------------------------
__device__ __forceinline__ void async_cp16(const _Float16* g, _Float16* l) {
#if __has_builtin(__builtin_amdgcn_global_load_async_to_lds_b128)
    __builtin_amdgcn_global_load_async_to_lds_b128(
        (__attribute__((address_space(1))) v4i*)(v4i*)g,
        (__attribute__((address_space(3))) v4i*)(v4i*)l, 0, 0);
#else
    *(uint4*)l = *(const uint4*)g;   // fallback: sync copy through regs
#endif
}

__device__ __forceinline__ void wait_async0() {
#if __has_builtin(__builtin_amdgcn_s_wait_asynccnt)
    __builtin_amdgcn_s_wait_asynccnt(0);
#elif __has_builtin(__builtin_amdgcn_global_load_async_to_lds_b128)
    asm volatile("s_wait_asynccnt 0x0" ::: "memory");
#endif
}

// ---------------------------------------------------------------------------
// WMMA fragment loaders (CDNA5 wave32 layouts, ISA 7.12.2)
// A 16x32 f16:  lane<16: row=lane, halves = K{0..7,16..23}
//               lane>=16: row=lane-16, halves = K{8..15,24..31}
// B 32x16 f16 (stored transposed in LDS, [n][k]):
//               lane<16: col=lane, halves = K{0..7,16..23}; lane>=16: K{8..15,24..31}
// ---------------------------------------------------------------------------
__device__ __forceinline__ v16h load_frag_a(const _Float16* base, int ld,
                                            int row0, int k0, int lane) {
    int r = row0 + (lane & 15);
    int koff = (lane >> 4) ? 8 : 0;
    const _Float16* p = base + r * ld + k0 + koff;
    union { v16h v; uint4 q[2]; } u;
    u.q[0] = *(const uint4*)p;        // K koff..koff+7
    u.q[1] = *(const uint4*)(p + 16); // K koff+16..koff+23
    return u.v;
}

__device__ __forceinline__ v16h load_frag_b(const _Float16* base, int ld,
                                            int col0, int k0, int lane) {
    int c = col0 + (lane & 15);
    int koff = (lane >> 4) ? 8 : 0;
    const _Float16* p = base + c * ld + k0 + koff;
    union { v16h v; uint4 q[2]; } u;
    u.q[0] = *(const uint4*)p;
    u.q[1] = *(const uint4*)(p + 16);
    return u.v;
}

// ---------------------------------------------------------------------------
// Generic fused WMMA GEMM:  C(MxN) = A(MxK,f16,rm) * Bt(NxK,f16,rm)^T
// Weights pre-transposed to column-major so all staging is contiguous 128-bit
// chunks. Double-buffered LDS tiles filled by ASYNC global->LDS DMA: tile k+1
// streams into buf^1 while tile k feeds the WMMA burst (8 per wave); then
// s_wait_asynccnt 0 + barrier. No VGPR staging -> no scratch spills.
// 128 thr = 4 waves, 2x2 waves of 32x32.
// M,N multiples of 64, K multiple of 64 (true for all shapes here).
// ---------------------------------------------------------------------------
__global__ __launch_bounds__(128, 1) void gemm_wmma_epi(
        const _Float16* __restrict__ A, const _Float16* __restrict__ Bt,
        int M, int N, int K, int mode, EpiArgs e)
{
    __shared__ _Float16 As[2][BM][LDT];
    __shared__ _Float16 Bs[2][BN][LDT];   // Bs[buf][n][k]

    const int tid  = threadIdx.x;
    const int lane = tid & 31;
    const int wave = tid >> 5;
    const int wm = wave >> 1, wn = wave & 1;
    const int row0 = blockIdx.y * BM;
    const int col0 = blockIdx.x * BN;

    // per-thread staging coordinates: 4 chunks of 8 halves for each operand
    const int cr  = tid >> 3;          // row/col within tile (0..15) + i*16
    const int ckc = (tid & 7) * 8;     // k offset within tile

    v8f acc[2][2] = {};

    // prologue: async-stage tile 0 into buf 0
    {
        unsigned aoff = (unsigned)(row0 + cr) * (unsigned)K + ckc;
        unsigned boff = (unsigned)(col0 + cr) * (unsigned)K + ckc;
#pragma unroll
        for (int i = 0; i < 4; ++i) {
            async_cp16(&A [aoff + (unsigned)(i * 16) * (unsigned)K], &As[0][cr + i * 16][ckc]);
            async_cp16(&Bt[boff + (unsigned)(i * 16) * (unsigned)K], &Bs[0][cr + i * 16][ckc]);
        }
    }
    wait_async0();
    __syncthreads();

    const int nk = K / BK;
    for (int kt = 0; kt < nk; ++kt) {
        const int cur = kt & 1, nxt = cur ^ 1;
        if (kt + 1 < nk) {  // async-stage next tile (overlaps the WMMA burst)
            unsigned kb   = (unsigned)((kt + 1) * BK);
            unsigned aoff = (unsigned)(row0 + cr) * (unsigned)K + kb + ckc;
            unsigned boff = (unsigned)(col0 + cr) * (unsigned)K + kb + ckc;
#pragma unroll
            for (int i = 0; i < 4; ++i) {
                async_cp16(&A [aoff + (unsigned)(i * 16) * (unsigned)K], &As[nxt][cr + i * 16][ckc]);
                async_cp16(&Bt[boff + (unsigned)(i * 16) * (unsigned)K], &Bs[nxt][cr + i * 16][ckc]);
            }
        }
#pragma unroll
        for (int ks = 0; ks < 2; ++ks) {
            v16h a0 = load_frag_a(&As[cur][0][0], LDT, wm * 32 + 0,  ks * 32, lane);
            v16h a1 = load_frag_a(&As[cur][0][0], LDT, wm * 32 + 16, ks * 32, lane);
            v16h b0 = load_frag_b(&Bs[cur][0][0], LDT, wn * 32 + 0,  ks * 32, lane);
            v16h b1 = load_frag_b(&Bs[cur][0][0], LDT, wn * 32 + 16, ks * 32, lane);
            acc[0][0] = __builtin_amdgcn_wmma_f32_16x16x32_f16(false, a0, false, b0, (short)0, acc[0][0], false, false);
            acc[0][1] = __builtin_amdgcn_wmma_f32_16x16x32_f16(false, a0, false, b1, (short)0, acc[0][1], false, false);
            acc[1][0] = __builtin_amdgcn_wmma_f32_16x16x32_f16(false, a1, false, b0, (short)0, acc[1][0], false, false);
            acc[1][1] = __builtin_amdgcn_wmma_f32_16x16x32_f16(false, a1, false, b1, (short)0, acc[1][1], false, false);
        }
        if (kt + 1 < nk) wait_async0();
        __syncthreads();
    }

    // Epilogue. C layout: lane<16 rows 0..7, lane>=16 rows 8..15; col = lane&15.
    const int rbase = (lane >> 4) ? 8 : 0;
#pragma unroll
    for (int mi = 0; mi < 2; ++mi) {
#pragma unroll
        for (int ni = 0; ni < 2; ++ni) {
            int colg = col0 + wn * 32 + ni * 16 + (lane & 15);
#pragma unroll
            for (int r = 0; r < 8; ++r) {
                int rowg = row0 + wm * 32 + mi * 16 + rbase + r;
                float v = acc[mi][ni][r];
                if (e.bias) v += e.bias[colg];
                size_t o = (size_t)rowg * N + colg;
                switch (mode) {
                    case EPI_F16:      e.outh[o] = (_Float16)v; break;
                    case EPI_GELU_F16: e.outh[o] = (_Float16)gelu_exact(v); break;
                    case EPI_STREAM: {
                        float w  = e.wsm[e.widx];
                        float vv = e.flag[rowg] ? e.empty[(size_t)(rowg & (SQ-1)) * N + colg] : v;
                        if (e.firstStream) e.outf[o] = w * vv;
                        else               e.outf[o] += w * vv;
                    } break;
                    case EPI_POS:
                        e.outf[o] = v + e.pos[(size_t)e.nidx[rowg] * N + colg];
                        break;
                    case EPI_QKV: {
                        int h = colg / DQ, d = colg % DQ;
                        int b = rowg >> 6, s = rowg & (SQ-1);
                        e.outh[(((size_t)b * HQ + h) * SQ + s) * DQ + d] = (_Float16)v;
                    } break;
                    case EPI_RES_H: e.outf[o] = v + (float)e.resh[o]; break;
                    case EPI_RES_F: e.outf[o] = v + e.resf[o]; break;
                }
            }
        }
    }
}

// ---------------------------------------------------------------------------
// Attention: one workgroup (128 thr, 4 waves) per (b,h).
// scores = Q K^T * 1/sqrt(48), query-row mask, softmax, Z = P V.
// ---------------------------------------------------------------------------
__global__ __launch_bounds__(128, 1) void attn_wmma(
        const _Float16* __restrict__ Q, const _Float16* __restrict__ Kk,
        const _Float16* __restrict__ V, const float* __restrict__ mask,
        _Float16* __restrict__ Z)
{
    __shared__ _Float16 Qs[SQ][72];      // D padded 48->64 (cols 48..63 zero)
    __shared__ _Float16 Ks[SQ][72];      // Ks[t][d] == B^T for scores
    __shared__ _Float16 Vt[DQ][72];      // Vt[d][t]  == B^T for Z
    __shared__ float    Sc[SQ][65];
    __shared__ _Float16 Pf[SQ][72];

    const int tid = threadIdx.x, lane = tid & 31, wave = tid >> 5;
    const int b = blockIdx.x >> 3, h = blockIdx.x & 7;
    const size_t base = (((size_t)b * HQ + h) * SQ) * DQ;

    for (int i = tid; i < SQ * DQ; i += 128) {
        int s = i / DQ, d = i % DQ;
        _Float16 q = Q[base + i], k = Kk[base + i], v = V[base + i];
        Qs[s][d] = q; Ks[s][d] = k; Vt[d][s] = v;
    }
    for (int i = tid; i < SQ * 16; i += 128) {       // zero pad d = 48..63
        int s = i / 16, d = 48 + (i % 16);
        Qs[s][d] = (_Float16)0.0f; Ks[s][d] = (_Float16)0.0f;
    }
    __syncthreads();

    // scores: wave handles query rows [16*wave, 16*wave+16)
    v8f sacc[4] = {};
#pragma unroll
    for (int k0 = 0; k0 < 64; k0 += 32) {
        v16h a = load_frag_a(&Qs[0][0], 72, wave * 16, k0, lane);
#pragma unroll
        for (int nt = 0; nt < 4; ++nt) {
            v16h bf = load_frag_b(&Ks[0][0], 72, nt * 16, k0, lane);
            sacc[nt] = __builtin_amdgcn_wmma_f32_16x16x32_f16(
                false, a, false, bf, (short)0, sacc[nt], false, false);
        }
    }
    const int rbase = (lane >> 4) ? 8 : 0;
    const float scale = 0.14433756729740643f;  // 1/sqrt(48)
#pragma unroll
    for (int nt = 0; nt < 4; ++nt)
#pragma unroll
        for (int r = 0; r < 8; ++r) {
            int s = wave * 16 + rbase + r;
            int t = nt * 16 + (lane & 15);
            float v = sacc[nt][r] * scale;
            if (mask[b * SQ + s] == 0.0f) v = -1.0e9f;   // mask4 broadcasts over t
            Sc[s][t] = v;
        }
    __syncthreads();

    if (tid < SQ) {  // row-wise softmax
        float m = -3.0e38f;
        for (int t = 0; t < SQ; ++t) m = fmaxf(m, Sc[tid][t]);
        float sum = 0.0f;
        for (int t = 0; t < SQ; ++t) { float ex = expf(Sc[tid][t] - m); Sc[tid][t] = ex; sum += ex; }
        float inv = 1.0f / sum;
        for (int t = 0; t < SQ; ++t) Pf[tid][t] = (_Float16)(Sc[tid][t] * inv);
    }
    __syncthreads();

    // Z = P(64x64) * V(64x48)
    v8f zacc[3] = {};
#pragma unroll
    for (int k0 = 0; k0 < 64; k0 += 32) {
        v16h a = load_frag_a(&Pf[0][0], 72, wave * 16, k0, lane);
#pragma unroll
        for (int nt = 0; nt < 3; ++nt) {
            v16h bf = load_frag_b(&Vt[0][0], 72, nt * 16, k0, lane);
            zacc[nt] = __builtin_amdgcn_wmma_f32_16x16x32_f16(
                false, a, false, bf, (short)0, zacc[nt], false, false);
        }
    }
#pragma unroll
    for (int nt = 0; nt < 3; ++nt)
#pragma unroll
        for (int r = 0; r < 8; ++r) {
            int s = wave * 16 + rbase + r;
            int d = nt * 16 + (lane & 15);
            Z[((size_t)b * SQ + s) * UQ + h * DQ + d] = (_Float16)zacc[nt][r];
        }
}

// ---------------------------------------------------------------------------
// LayerNorm over 384 cols: one block (128 thr) per row; writes f16.
// ---------------------------------------------------------------------------
__global__ __launch_bounds__(128) void ln_kernel(
        const float* __restrict__ X, const float* __restrict__ g,
        const float* __restrict__ bb, _Float16* __restrict__ Y)
{
    const int row = blockIdx.x;
    const float* x = X + (size_t)row * UQ;
    float v[3], s = 0.f, s2 = 0.f;
#pragma unroll
    for (int i = 0; i < 3; ++i) {
        v[i] = x[threadIdx.x + i * 128];
        s += v[i]; s2 += v[i] * v[i];
    }
    __shared__ float r1[128], r2[128];
    r1[threadIdx.x] = s; r2[threadIdx.x] = s2; __syncthreads();
    for (int off = 64; off > 0; off >>= 1) {
        if (threadIdx.x < off) { r1[threadIdx.x] += r1[threadIdx.x + off];
                                 r2[threadIdx.x] += r2[threadIdx.x + off]; }
        __syncthreads();
    }
    float mean = r1[0] * (1.0f / UQ);
    float var  = r2[0] * (1.0f / UQ) - mean * mean;
    float rstd = rsqrtf(var + 1e-6f);
    _Float16* y = Y + (size_t)row * UQ;
#pragma unroll
    for (int i = 0; i < 3; ++i) {
        int c = threadIdx.x + i * 128;
        y[c] = (_Float16)((v[i] - mean) * rstd * g[c] + bb[c]);
    }
}

// ---------------------------------------------------------------------------
// Small helper kernels (fp32 paths for tiny GEMMs / glue)
// ---------------------------------------------------------------------------
__global__ void embed_gemm_gelu(const float* __restrict__ X, const float* __restrict__ W,
                                _Float16* __restrict__ T, int K)
{
    int idx = blockIdx.x * 256 + threadIdx.x;       // M*384 threads exactly
    int row = idx / UQ, col = idx % UQ;
    const float* x = X + (size_t)row * K;
    float acc = 0.f;
    for (int k = 0; k < K; ++k) acc += x[k] * W[(size_t)k * UQ + col];
    T[idx] = (_Float16)gelu_exact(acc);
}

__global__ void flags_kernel(const float* lips, const float* lh, const float* rh,
                             const float* po, unsigned char* f)
{
    int row = blockIdx.x * 256 + threadIdx.x;
    if (row >= MQ) return;
    float s;
    s = 0; for (int k = 0; k < 80; ++k) s += lips[(size_t)row * 80 + k]; f[row]          = (s == 0.f);
    s = 0; for (int k = 0; k < 42; ++k) s += lh[(size_t)row * 42 + k];   f[MQ + row]     = (s == 0.f);
    s = 0; for (int k = 0; k < 42; ++k) s += rh[(size_t)row * 42 + k];   f[2 * MQ + row] = (s == 0.f);
    s = 0; for (int k = 0; k < 20; ++k) s += po[(size_t)row * 20 + k];   f[3 * MQ + row] = (s == 0.f);
}

__global__ void softmax4_kernel(const float* lw, float* w) {
    if (threadIdx.x == 0) {
        float m = fmaxf(fmaxf(lw[0], lw[1]), fmaxf(lw[2], lw[3]));
        float e0 = expf(lw[0] - m), e1 = expf(lw[1] - m),
              e2 = expf(lw[2] - m), e3 = expf(lw[3] - m);
        float s = e0 + e1 + e2 + e3;
        w[0] = e0 / s; w[1] = e1 / s; w[2] = e2 / s; w[3] = e3 / s;
    }
}

__global__ void normidx_kernel(const float* fr, int* nidx) {
    int b = blockIdx.x, t = threadIdx.x;  // 64 threads
    __shared__ float red[64];
    float v = fr[b * SQ + t];
    red[t] = v; __syncthreads();
    for (int off = 32; off > 0; off >>= 1) {
        if (t < off) red[t] = fmaxf(red[t], red[t + off]);
        __syncthreads();
    }
    float mx = red[0];
    int idx = (v == -1.0f) ? SQ : (int)(floorf(v / mx) * (float)SQ);
    nidx[b * SQ + t] = idx;
}

__global__ void pool_kernel(const float* X, const float* mask, float* pooled) {
    int b = blockIdx.x, c = threadIdx.x;  // 384 threads
    float s = 0.f, dn = 0.f;
    for (int t = 0; t < SQ; ++t) {
        float m = mask[b * SQ + t];
        s += X[((size_t)b * SQ + t) * UQ + c] * m;
        dn += m;
    }
    pooled[(size_t)b * UQ + c] = s / dn;
}

__global__ void outproj_kernel(const float* pooled, const float* W,
                               const float* bias, float* out) {
    int idx = blockIdx.x * 256 + threadIdx.x;
    if (idx >= BQ * NCQ) return;
    int b = idx / NCQ, c = idx % NCQ;
    float a = bias[c];
    for (int u = 0; u < UQ; ++u) a += pooled[(size_t)b * UQ + u] * W[(size_t)u * NCQ + c];
    out[idx] = a;
}

__global__ void cvt_f16_kernel(const float* s, _Float16* d, int n) {
    int i = blockIdx.x * 256 + threadIdx.x;
    if (i < n) d[i] = (_Float16)s[i];
}

// row-major (K,N) f32 -> column-major (N,K) f16  (coalesced read)
__global__ void cvt_f16_t_kernel(const float* s, _Float16* d, int K, int N) {
    int i = blockIdx.x * 256 + threadIdx.x;
    if (i >= K * N) return;
    int k = i / N, n = i % N;
    d[(size_t)n * K + k] = (_Float16)s[i];
}

// (H,U,D) -> column-major (H*D, U): d[c*U + u] = s[(h*U+u)*D + dd], c = h*D+dd
__global__ void cvt_qkv_t_kernel(const float* s, _Float16* d) {
    int i = blockIdx.x * 256 + threadIdx.x;
    if (i >= UQ * UQ) return;
    int u = i / UQ, c = i % UQ;
    int h = c / DQ, dd = c % DQ;
    d[(size_t)c * UQ + u] = (_Float16)s[((size_t)h * UQ + u) * DQ + dd];
}

// ---------------------------------------------------------------------------
// Host orchestration
// ---------------------------------------------------------------------------
extern "C" void kernel_launch(void* const* d_in, const int* in_sizes, int n_in,
                              void* d_out, int out_size, void* d_ws, size_t ws_size,
                              hipStream_t stream) {
    (void)in_sizes; (void)n_in; (void)out_size; (void)ws_size;

    auto F = [&](int i) { return (const float*)d_in[i]; };
    const float* lips = F(0); const float* lhx = F(1);
    const float* rhx = F(2);  const float* pox = F(3);
    const float* fridx = F(4); const float* amask = F(5);
    const int IW1[4] = {6, 9, 12, 15};
    const int IW2[4] = {7, 10, 13, 16};
    const int IEM[4] = {8, 11, 14, 17};
    const int KIN[4] = {80, 42, 42, 20};
    const float* inx[4] = {lips, lhx, rhx, pox};
    const float* lw = F(18);
    const float* fc_w1 = F(19); const float* fc_w2 = F(20);
    const float* pos = F(21);
    const float* out_w = F(22); const float* out_b = F(23);
    const int PB = 24;  // blocks base; 16 leaves per block

    char* wsp = (char*)d_ws; size_t off = 0;
    auto alloc = [&](size_t bytes) {
        void* p = wsp + off;
        off = (off + bytes + 255) & ~(size_t)255;
        return p;
    };
    float*     wsm   = (float*)alloc(4 * sizeof(float));
    unsigned char* flg = (unsigned char*)alloc(4 * MQ);
    int*       nidx  = (int*)alloc(MQ * sizeof(int));
    _Float16*  T     = (_Float16*)alloc((size_t)MQ * UQ * 2);
    float*     X     = (float*)alloc((size_t)MQ * UQ * 4);
    _Float16*  Xh    = (_Float16*)alloc((size_t)MQ * UQ * 2);
    _Float16*  X1h   = (_Float16*)alloc((size_t)MQ * UQ * 2);
    _Float16*  Qb    = (_Float16*)alloc((size_t)MQ * UQ * 2);
    _Float16*  Kb    = (_Float16*)alloc((size_t)MQ * UQ * 2);
    _Float16*  Vb    = (_Float16*)alloc((size_t)MQ * UQ * 2);
    _Float16*  Zb    = (_Float16*)alloc((size_t)MQ * UQ * 2);
    float*     X2    = (float*)alloc((size_t)MQ * UQ * 4);
    _Float16*  X3h   = (_Float16*)alloc((size_t)MQ * UQ * 2);
    _Float16*  G1h   = (_Float16*)alloc((size_t)MQ * MLPQ * 2);
    float*     pooled = (float*)alloc((size_t)BQ * UQ * 4);
    _Float16*  w2h[4]; for (int i = 0; i < 4; ++i) w2h[i] = (_Float16*)alloc((size_t)UQ * UQ * 2);
    _Float16*  fc1h = (_Float16*)alloc((size_t)UQ * UQ * 2);
    _Float16*  fc2h = (_Float16*)alloc((size_t)UQ * UQ * 2);
    _Float16 *wqh[4], *wkh[4], *wvh[4], *woh[4], *mw1h[4], *mw2h[4];
    for (int b = 0; b < 4; ++b) {
        wqh[b]  = (_Float16*)alloc((size_t)UQ * UQ * 2);
        wkh[b]  = (_Float16*)alloc((size_t)UQ * UQ * 2);
        wvh[b]  = (_Float16*)alloc((size_t)UQ * UQ * 2);
        woh[b]  = (_Float16*)alloc((size_t)UQ * UQ * 2);
        mw1h[b] = (_Float16*)alloc((size_t)UQ * MLPQ * 2);
        mw2h[b] = (_Float16*)alloc((size_t)MLPQ * UQ * 2);
    }

    auto epi0 = []() { EpiArgs e{}; return e; };
    const int gW  = (UQ * UQ + 255) / 256;
    const int gW2 = (UQ * MLPQ + 255) / 256;
    const int gMU = (MQ * UQ + 255) / 256;
    dim3 gemmGrid(UQ / BN, MQ / BM);                // N=384
    dim3 gemmGridW(MLPQ / BN, MQ / BM);             // N=768

    // -------- scalars / glue --------
    softmax4_kernel<<<1, 64, 0, stream>>>(lw, wsm);
    flags_kernel<<<MQ / 256, 256, 0, stream>>>(lips, lhx, rhx, pox, flg);
    normidx_kernel<<<BQ, 64, 0, stream>>>(fridx, nidx);

    // -------- weight conversions (f32 row-major -> f16 column-major) --------
    for (int i = 0; i < 4; ++i)
        cvt_f16_t_kernel<<<gW, 256, 0, stream>>>(F(IW2[i]), w2h[i], UQ, UQ);
    cvt_f16_t_kernel<<<gW, 256, 0, stream>>>(fc_w1, fc1h, UQ, UQ);
    cvt_f16_t_kernel<<<gW, 256, 0, stream>>>(fc_w2, fc2h, UQ, UQ);
    for (int b = 0; b < 4; ++b) {
        int pb = PB + 16 * b;
        cvt_qkv_t_kernel<<<gW, 256, 0, stream>>>(F(pb + 2), wqh[b]);
        cvt_qkv_t_kernel<<<gW, 256, 0, stream>>>(F(pb + 4), wkh[b]);
        cvt_qkv_t_kernel<<<gW, 256, 0, stream>>>(F(pb + 6), wvh[b]);
        cvt_f16_t_kernel<<<gW, 256, 0, stream>>>(F(pb + 8), woh[b], UQ, UQ);
        cvt_f16_t_kernel<<<gW2, 256, 0, stream>>>(F(pb + 12), mw1h[b], UQ, MLPQ);
        cvt_f16_t_kernel<<<gW2, 256, 0, stream>>>(F(pb + 14), mw2h[b], MLPQ, UQ);
    }

    // -------- landmark streams --------
    for (int i = 0; i < 4; ++i) {
        embed_gemm_gelu<<<(MQ * UQ) / 256, 256, 0, stream>>>(inx[i], F(IW1[i]), T, KIN[i]);
        EpiArgs e = epi0();
        e.outf = X; e.wsm = wsm; e.widx = i; e.flag = flg + (size_t)i * MQ;
        e.empty = F(IEM[i]); e.firstStream = (i == 0);
        gemm_wmma_epi<<<gemmGrid, 128, 0, stream>>>(T, w2h[i], MQ, UQ, UQ, EPI_STREAM, e);
    }

    // -------- fc + positional --------
    cvt_f16_kernel<<<gMU, 256, 0, stream>>>(X, Xh, MQ * UQ);
    { EpiArgs e = epi0(); e.outh = T;
      gemm_wmma_epi<<<gemmGrid, 128, 0, stream>>>(Xh, fc1h, MQ, UQ, UQ, EPI_GELU_F16, e); }
    { EpiArgs e = epi0(); e.outf = X; e.nidx = nidx; e.pos = pos;
      gemm_wmma_epi<<<gemmGrid, 128, 0, stream>>>(T, fc2h, MQ, UQ, UQ, EPI_POS, e); }

    // -------- transformer blocks --------
    for (int b = 0; b < 4; ++b) {
        int pb = PB + 16 * b;
        ln_kernel<<<MQ, 128, 0, stream>>>(X, F(pb + 0), F(pb + 1), X1h);
        { EpiArgs e = epi0(); e.outh = Qb; e.bias = F(pb + 3);
          gemm_wmma_epi<<<gemmGrid, 128, 0, stream>>>(X1h, wqh[b], MQ, UQ, UQ, EPI_QKV, e); }
        { EpiArgs e = epi0(); e.outh = Kb; e.bias = F(pb + 5);
          gemm_wmma_epi<<<gemmGrid, 128, 0, stream>>>(X1h, wkh[b], MQ, UQ, UQ, EPI_QKV, e); }
        { EpiArgs e = epi0(); e.outh = Vb; e.bias = F(pb + 7);
          gemm_wmma_epi<<<gemmGrid, 128, 0, stream>>>(X1h, wvh[b], MQ, UQ, UQ, EPI_QKV, e); }
        attn_wmma<<<BQ * HQ, 128, 0, stream>>>(Qb, Kb, Vb, amask, Zb);
        { EpiArgs e = epi0(); e.outf = X2; e.bias = F(pb + 9); e.resh = X1h;
          gemm_wmma_epi<<<gemmGrid, 128, 0, stream>>>(Zb, woh[b], MQ, UQ, UQ, EPI_RES_H, e); }
        ln_kernel<<<MQ, 128, 0, stream>>>(X2, F(pb + 10), F(pb + 11), X3h);
        { EpiArgs e = epi0(); e.outh = G1h; e.bias = F(pb + 13);
          gemm_wmma_epi<<<gemmGridW, 128, 0, stream>>>(X3h, mw1h[b], MQ, MLPQ, UQ, EPI_GELU_F16, e); }
        { EpiArgs e = epi0(); e.outf = X; e.bias = F(pb + 15); e.resf = X2;
          gemm_wmma_epi<<<gemmGrid, 128, 0, stream>>>(G1h, mw2h[b], MQ, UQ, MLPQ, EPI_RES_F, e); }
    }

    // -------- pool + classifier head --------
    pool_kernel<<<BQ, UQ, 0, stream>>>(X, amask, pooled);
    outproj_kernel<<<(BQ * NCQ + 255) / 256, 256, 0, stream>>>(pooled, out_w, out_b,
                                                               (float*)d_out);
}
